// LSTMActorCritic_2705829397327
// MI455X (gfx1250) — matrix operand order repaired and checked by reading, not compiled, not run on hardware
//
#include <hip/hip_runtime.h>
#include <hip/hip_bf16.h>

// ---------------------------------------------------------------------------
// Problem sizes (fixed by the reference)
// ---------------------------------------------------------------------------
#define TT   512
#define BB   256
#define OBS  256
#define HID  512
#define LSTMD 512
#define ACTD 32
#define G4   (4 * LSTMD)   // 2048
#define MTB  (TT * BB)     // 131072 rows for the big GEMMs

// ---------------------------------------------------------------------------
// Types
// ---------------------------------------------------------------------------
typedef __attribute__((ext_vector_type(16))) __bf16 v16bf;
typedef __attribute__((ext_vector_type(8)))  float  v8f;
typedef unsigned int u32x4 __attribute__((ext_vector_type(4)));

// global-address-space alias to force global_load/global_store (never FLAT)
#define AS1 __attribute__((address_space(1)))

// exact pointer types for __builtin_amdgcn_global_load_async_to_lds_b128
// (param 0: int4 addrspace(1)*, param 1: int4 addrspace(3)*, per hipcc diag)
typedef int i32x4v __attribute__((vector_size(16)));
typedef AS1 i32x4v* gv4_ptr;
typedef __attribute__((address_space(3))) i32x4v* lv4_ptr;

// gfx1250 async global->LDS path (ASYNCcnt). Fall back to plain LDS copies if
// the builtin is not available in this toolchain.
#if __has_builtin(__builtin_amdgcn_global_load_async_to_lds_b128)
#define HAVE_ASYNC_LDS 1
#else
#define HAVE_ASYNC_LDS 0
#endif

// ---------------------------------------------------------------------------
// Device helpers
// ---------------------------------------------------------------------------
__device__ __forceinline__ float sigmoidf_(float x) {
    return 1.0f / (1.0f + expf(-x));
}

__device__ __forceinline__ float gelu_tanh_(float x) {
    const float k0 = 0.7978845608028654f;  // sqrt(2/pi)
    const float k1 = 0.044715f;
    float t = tanhf(k0 * (x + k1 * x * x * x));
    return 0.5f * x * (1.0f + t);
}

// A-matrix fragment (16x32 bf16, MxK) for v_wmma_f32_16x16x32_bf16.
// Lane L holds row M = L%16; K = kb..kb+15,
// kb = 16*(L/16). Row-major A -> two contiguous 16B global loads per lane.
__device__ __forceinline__ v16bf load_a_frag(const __bf16* __restrict__ A, int lda,
                                             int row0, int k0, int lane) {
    int r  = row0 + (lane & 15);
    int kb = k0 + ((lane >> 4) << 4);
    const AS1 u32x4* p = (const AS1 u32x4*)(A + (size_t)r * lda + kb);
    union { v16bf v; u32x4 q[2]; } u;
    u.q[0] = p[0];
    u.q[1] = p[1];
    return u.v;
}

// B-matrix fragment (32x16 bf16, KxN), pre-packed so each lane's 16 values
// (col N = lane%16, K = 16*(lane/16) + 0..15) are 32 contiguous bytes.
__device__ __forceinline__ v16bf load_b_frag(const __bf16* __restrict__ Bp,
                                             int ktiles, int ntile, int ktile, int lane) {
    const AS1 u32x4* p =
        (const AS1 u32x4*)(Bp + (((size_t)ntile * ktiles + ktile) * 32 + lane) * 16);
    union { v16bf v; u32x4 q[2]; } u;
    u.q[0] = p[0];
    u.q[1] = p[1];
    return u.v;
}

__device__ __forceinline__ v8f wmma_bf16(v16bf a, v16bf b, v8f c) {
    // D = A(16x32) * B(32x16) + C, f32 accumulate
    return __builtin_amdgcn_wmma_f32_16x16x32_bf16(
        /*neg_a=*/false, a, /*neg_b=*/false, b,
        /*c_mod=*/(short)0, c, /*reuse_a=*/false, /*reuse_b=*/false);
}

// ---------------------------------------------------------------------------
// Weight pre-pack: f32 row-major (K x N) -> bf16 per-lane B fragments
// tile = ntile*ktiles + ktile ; each (tile, lane) owns 16 contiguous bf16.
// ---------------------------------------------------------------------------
__global__ void pack_w_kernel(const float* __restrict__ W, __bf16* __restrict__ Wp,
                              int K, int N) {
    int t    = blockIdx.x * blockDim.x + threadIdx.x;
    int lane = t & 31;
    int tile = t >> 5;
    int ktiles = K >> 5;
    int ntiles = N >> 4;
    if (tile >= ktiles * ntiles) return;
    int ktile = tile % ktiles;
    int ntile = tile / ktiles;
    int n  = ntile * 16 + (lane & 15);
    int kb = ktile * 32 + ((lane >> 4) << 4);   // 16 consecutive K per lane
    AS1 __bf16* dst = (AS1 __bf16*)(Wp + ((size_t)tile * 32 + lane) * 16);
    const AS1 float* src = (const AS1 float*)W;
#pragma unroll
    for (int i = 0; i < 16; ++i)
        dst[i] = (__bf16)src[(size_t)(kb + i) * N + n];
}

__global__ void f32_to_bf16_kernel(const float* __restrict__ s, __bf16* __restrict__ d, int n) {
    int i = blockIdx.x * blockDim.x + threadIdx.x;
    if (i < n) ((AS1 __bf16*)d)[i] = (__bf16)((const AS1 float*)s)[i];
}

__global__ void zero_u32_kernel(unsigned int* __restrict__ p, int n) {
    int i = blockIdx.x * blockDim.x + threadIdx.x;
    if (i < n) ((AS1 unsigned int*)p)[i] = 0u;
}

// ---------------------------------------------------------------------------
// Generic WMMA GEMM: out = act(A(bf16, MxK row-major) @ Bpacked + bias)
//   block tile = (WM*16) x (WN*16), one wmma accumulator per wave
// ---------------------------------------------------------------------------
template <int WM, int WN, bool GELU, bool OUT_BF16>
__global__ __launch_bounds__(WM * WN * 32)
void gemm_bf16_kernel(const __bf16* __restrict__ A, int lda,
                      const __bf16* __restrict__ Bp,
                      const float* __restrict__ bias,
                      void* __restrict__ out, int N, int K) {
    int lane = threadIdx.x & 31;
    int wave = threadIdx.x >> 5;
    int wm = wave / WN, wn = wave % WN;
    int row0 = blockIdx.x * (WM * 16) + wm * 16;
    int col0 = blockIdx.y * (WN * 16) + wn * 16;
    int ktiles = K >> 5;
    int ntile  = col0 >> 4;

    v8f acc = {};
#pragma unroll 4
    for (int kt = 0; kt < ktiles; ++kt) {
        v16bf a = load_a_frag(A, lda, row0, kt * 32, lane);
        v16bf b = load_b_frag(Bp, ktiles, ntile, kt, lane);
        acc = wmma_bf16(a, b, acc);
    }

    // C/D layout: element r of v8f -> M = r + 8*(lane/16), N = lane%16
    int n  = lane & 15;
    int mh = (lane >> 4) << 3;
    int col = col0 + n;
    float bv = ((const AS1 float*)bias)[col];
#pragma unroll
    for (int r = 0; r < 8; ++r) {
        int row = row0 + r + mh;
        float z = acc[r] + bv;
        if (GELU) z = gelu_tanh_(z);
        if (OUT_BF16)
            ((AS1 __bf16*)out)[(size_t)row * N + col] = (__bf16)z;
        else
            ((AS1 float*)out)[(size_t)row * N + col] = z;
    }
}

// ---------------------------------------------------------------------------
// One LSTM time step (launched T times, sequential on the stream):
//   z = feats[t] @ W_i + h_prev @ W_h + b_lstm        (fused, K = 512 + 512)
//   i,f,g,o gates -> c, h ; h written to h_next and ys[t] (bf16)
//
// Block: 256 thr = 8 waves (2x4), tile 32 rows(b) x 64 cols(j), 4 gate accs.
// A panels (feats[t] 32x512, h_prev 32x512) staged in LDS (async-to-LDS when
// available), padded stride 520 to spread banks. Fully unrolled 32-tile K
// loop with software-pipelined B fragments (statically selected Wi/Wh).
// Grid: (B/32, LSTMD/64) = (8, 8)
// ---------------------------------------------------------------------------
#define PSTR 520   // padded LDS row stride in bf16 elements (1040 B)

__global__ __launch_bounds__(256)
void lstm_step_kernel(const __bf16* __restrict__ feats_t,
                      const __bf16* __restrict__ h_prev,
                      const __bf16* __restrict__ Wi_p,
                      const __bf16* __restrict__ Wh_p,
                      const float* __restrict__ b_lstm,
                      float* __restrict__ c,
                      __bf16* __restrict__ h_next,
                      __bf16* __restrict__ ys_t) {
    constexpr int KT = HID >> 5;          // 16 K-tiles per 512-deep GEMM
    __shared__ __bf16 smem[2][32][PSTR];  // [panel][row][k], ~65 KB

    int tid  = threadIdx.x;
    int lane = tid & 31;
    int wave = tid >> 5;
    int wm = wave >> 2, wn = wave & 3;
    int rowblk = blockIdx.x * 32;                  // batch-row base of block
    int col0   = blockIdx.y * 64 + wn * 16;        // j within LSTMD
    int ntile  = col0 >> 4;                        // 0..31 within one gate

    // ---- stage the two A panels into LDS (coalesced 128B per thread/panel)
    {
        int lrow = tid >> 3;            // 0..31
        int seg  = (tid & 7) * 64;      // 64 bf16 = 128 B per segment
        const __bf16* g0 = feats_t + (size_t)(rowblk + lrow) * HID   + seg;
        const __bf16* g1 = h_prev  + (size_t)(rowblk + lrow) * LSTMD + seg;
        __bf16* l0 = &smem[0][lrow][seg];
        __bf16* l1 = &smem[1][lrow][seg];
#if HAVE_ASYNC_LDS
#pragma unroll
        for (int i = 0; i < 8; ++i)
            __builtin_amdgcn_global_load_async_to_lds_b128(
                (gv4_ptr)(g0 + i * 8), (lv4_ptr)(l0 + i * 8), 0, 0);
#pragma unroll
        for (int i = 0; i < 8; ++i)
            __builtin_amdgcn_global_load_async_to_lds_b128(
                (gv4_ptr)(g1 + i * 8), (lv4_ptr)(l1 + i * 8), 0, 0);
        asm volatile("s_wait_asynccnt 0x0" ::: "memory");
#else
#pragma unroll
        for (int i = 0; i < 8; ++i)
            *(u32x4*)(l0 + i * 8) = ((const AS1 u32x4*)(g0))[i];
#pragma unroll
        for (int i = 0; i < 8; ++i)
            *(u32x4*)(l1 + i * 8) = ((const AS1 u32x4*)(g1))[i];
#endif
        __syncthreads();
    }

    // ---- fully unrolled 32-tile K loop with pipelined B fragments ----
    v8f acc[4] = {};                               // i, f, g, o
    v16bf bcur[4];
#pragma unroll
    for (int g = 0; g < 4; ++g)
        bcur[g] = load_b_frag(Wi_p, KT, g * 32 + ntile, 0, lane);

    int arow = (wm << 4) + (lane & 15);            // LDS row for this lane
#pragma unroll
    for (int kt = 0; kt < 32; ++kt) {
        // A fragment from LDS (two 16B ds loads)
        const int kb = ((kt & 15) << 5) + ((lane >> 4) << 4);
        const __bf16* prow = &smem[kt >> 4][arow][kb];
        union { v16bf v; u32x4 q[2]; } ua;
        ua.q[0] = *(const u32x4*)(prow);
        ua.q[1] = *(const u32x4*)(prow + 8);

        // prefetch next k-tile's B fragments (Wi/Wh resolved at compile time)
        const int ktn = kt + 1;
        v16bf bnext[4];
        if (ktn < 32) {
            const __bf16* W = (ktn < 16) ? Wi_p : Wh_p;
#pragma unroll
            for (int g = 0; g < 4; ++g)
                bnext[g] = load_b_frag(W, KT, g * 32 + ntile, ktn & 15, lane);
        }
#pragma unroll
        for (int g = 0; g < 4; ++g) acc[g] = wmma_bf16(ua.v, bcur[g], acc[g]);
#pragma unroll
        for (int g = 0; g < 4; ++g)
            if (ktn < 32) bcur[g] = bnext[g];
    }

    // ---- LSTM elementwise epilogue (global loads/stores, addrspace(1)) ----
    int n  = lane & 15;
    int mh = (lane >> 4) << 3;
    int j  = col0 + n;
    const AS1 float* bl = (const AS1 float*)b_lstm;
    AS1 float*  cg = (AS1 float*)c;
    AS1 __bf16* hg = (AS1 __bf16*)h_next;
    AS1 __bf16* yg = (AS1 __bf16*)ys_t;
    float bi = bl[0 * LSTMD + j];
    float bf = bl[1 * LSTMD + j];
    float bg = bl[2 * LSTMD + j];
    float bo = bl[3 * LSTMD + j];
    int row0 = rowblk + (wm << 4);
#pragma unroll
    for (int r = 0; r < 8; ++r) {
        int row = row0 + r + mh;
        size_t idx = (size_t)row * LSTMD + j;
        float zi = acc[0][r] + bi;
        float zf = acc[1][r] + bf;
        float zg = acc[2][r] + bg;
        float zo = acc[3][r] + bo;
        float ig = sigmoidf_(zi);
        float fg = sigmoidf_(zf);
        float gg = tanhf(zg);
        float og = sigmoidf_(zo);
        float cn = fg * cg[idx] + ig * gg;
        float hn = og * tanhf(cn);
        cg[idx] = cn;
        __bf16 hb = (__bf16)hn;
        hg[idx] = hb;
        yg[idx] = hb;
    }
}

// ---------------------------------------------------------------------------
// sigma = exp(log_std) broadcast over (T, B, ACT)
// ---------------------------------------------------------------------------
__global__ void sigma_kernel(const float* __restrict__ log_std, float* __restrict__ out, int n) {
    int i = blockIdx.x * blockDim.x + threadIdx.x;
    if (i < n) ((AS1 float*)out)[i] = expf(((const AS1 float*)log_std)[i & (ACTD - 1)]);
}

// ---------------------------------------------------------------------------
// value = ys @ W_v + b_v  (N == 1 -> per-row dot, f32 accumulate)
// ---------------------------------------------------------------------------
__global__ void value_kernel(const __bf16* __restrict__ ys, const float* __restrict__ Wv,
                             const float* __restrict__ bv, float* __restrict__ out, int rows) {
    int r = blockIdx.x * blockDim.x + threadIdx.x;
    if (r >= rows) return;
    const AS1 u32x4* p = (const AS1 u32x4*)(ys + (size_t)r * LSTMD);
    const AS1 float* wv = (const AS1 float*)Wv;
    float s = 0.0f;
    for (int k = 0; k < LSTMD; k += 8) {
        union { u32x4 q; __bf16 e[8]; } u;
        u.q = p[k >> 3];
#pragma unroll
        for (int i = 0; i < 8; ++i) s += (float)u.e[i] * wv[k + i];
    }
    ((AS1 float*)out)[r] = s + ((const AS1 float*)bv)[0];
}

// ---------------------------------------------------------------------------
// Host-side launcher
// ---------------------------------------------------------------------------
extern "C" void kernel_launch(void* const* d_in, const int* in_sizes, int n_in,
                              void* d_out, int out_size, void* d_ws, size_t ws_size,
                              hipStream_t stream) {
    (void)in_sizes; (void)n_in; (void)out_size; (void)ws_size;

    const float* x       = (const float*)d_in[0];   // (T,B,OBS)
    const float* W_enc   = (const float*)d_in[1];   // (OBS,HID)
    const float* b_enc   = (const float*)d_in[2];   // (HID)
    const float* W_i     = (const float*)d_in[3];   // (HID,4H)
    const float* W_h     = (const float*)d_in[4];   // (H,4H)
    const float* b_lstm  = (const float*)d_in[5];   // (4H)
    const float* W_mu    = (const float*)d_in[6];   // (H,ACT)
    const float* b_mu    = (const float*)d_in[7];   // (ACT)
    const float* log_std = (const float*)d_in[8];   // (1,ACT)
    const float* W_v     = (const float*)d_in[9];   // (H,1)
    const float* b_v     = (const float*)d_in[10];  // (1)

    float* mu_out    = (float*)d_out;                       // T*B*ACT
    float* sigma_out = mu_out + (size_t)MTB * ACTD;         // T*B*ACT
    float* val_out   = sigma_out + (size_t)MTB * ACTD;      // T*B

    // ---- workspace carve (256B aligned) ----
    char* w = (char*)d_ws;
    auto carve = [&](size_t bytes) -> void* {
        void* p = (void*)w;
        w += (bytes + 255) & ~(size_t)255;
        return p;
    };
    __bf16* xb     = (__bf16*)carve((size_t)MTB * OBS * 2);      //  64 MB
    __bf16* feats  = (__bf16*)carve((size_t)MTB * HID * 2);      // 128 MB
    __bf16* ys     = (__bf16*)carve((size_t)MTB * LSTMD * 2);    // 128 MB
    __bf16* wenc_p = (__bf16*)carve((size_t)OBS * HID * 2);
    __bf16* wi_p   = (__bf16*)carve((size_t)HID * G4 * 2);
    __bf16* wh_p   = (__bf16*)carve((size_t)LSTMD * G4 * 2);
    __bf16* wmu_p  = (__bf16*)carve((size_t)LSTMD * ACTD * 2);
    __bf16* h0     = (__bf16*)carve((size_t)BB * LSTMD * 2);
    __bf16* h1     = (__bf16*)carve((size_t)BB * LSTMD * 2);
    float*  cbuf   = (float*)carve((size_t)BB * LSTMD * 4);

    // ---- one-time (per launch) weight packs + input conversion ----
    {
        int thr;
        thr = (HID / 16) * (OBS / 32) * 32;      // W_enc
        pack_w_kernel<<<(thr + 255) / 256, 256, 0, stream>>>(W_enc, wenc_p, OBS, HID);
        thr = (G4 / 16) * (HID / 32) * 32;       // W_i
        pack_w_kernel<<<(thr + 255) / 256, 256, 0, stream>>>(W_i, wi_p, HID, G4);
        thr = (G4 / 16) * (LSTMD / 32) * 32;     // W_h
        pack_w_kernel<<<(thr + 255) / 256, 256, 0, stream>>>(W_h, wh_p, LSTMD, G4);
        thr = (ACTD / 16) * (LSTMD / 32) * 32;   // W_mu
        pack_w_kernel<<<(thr + 255) / 256, 256, 0, stream>>>(W_mu, wmu_p, LSTMD, ACTD);

        int nx = MTB * OBS;
        f32_to_bf16_kernel<<<(nx + 255) / 256, 256, 0, stream>>>(x, xb, nx);

        int nh = (BB * LSTMD * 2) / 4;           // h0 (bf16) in u32 words
        zero_u32_kernel<<<(nh + 255) / 256, 256, 0, stream>>>((unsigned int*)h0, nh);
        int nc = BB * LSTMD;                     // c (f32) in u32 words
        zero_u32_kernel<<<(nc + 255) / 256, 256, 0, stream>>>((unsigned int*)cbuf, nc);
    }

    // ---- encoder: feats = gelu(xb @ W_enc + b_enc), bf16 out ----
    gemm_bf16_kernel<2, 4, true, true>
        <<<dim3(MTB / 32, HID / 64), 256, 0, stream>>>(
            xb, OBS, wenc_p, b_enc, (void*)feats, HID, OBS);

    // ---- recurrence: 512 sequential steps, h double-buffered ----
    for (int t = 0; t < TT; ++t) {
        const __bf16* hp = (t & 1) ? h1 : h0;
        __bf16*       hn = (t & 1) ? h0 : h1;
        lstm_step_kernel<<<dim3(BB / 32, LSTMD / 64), 256, 0, stream>>>(
            feats + (size_t)t * BB * HID, hp, wi_p, wh_p, b_lstm,
            cbuf, hn, ys + (size_t)t * BB * LSTMD);
    }

    // ---- heads ----
    gemm_bf16_kernel<2, 2, false, false>
        <<<dim3(MTB / 32, ACTD / 32), 128, 0, stream>>>(
            ys, LSTMD, wmu_p, b_mu, (void*)mu_out, ACTD, LSTMD);

    int ns = MTB * ACTD;
    sigma_kernel<<<(ns + 255) / 256, 256, 0, stream>>>(log_std, sigma_out, ns);

    value_kernel<<<(MTB + 255) / 256, 256, 0, stream>>>(ys, W_v, b_v, val_out, MTB);
}
